// LinearSelfAttention_65738769433262
// MI455X (gfx1250) — matrix operand order repaired and checked by reading, not compiled, not run on hardware
//
#include <hip/hip_runtime.h>
#include <hip/hip_bf16.h>

typedef __attribute__((ext_vector_type(16))) _Float16 v16h;
typedef __attribute__((ext_vector_type(8)))  _Float16 v8h;
typedef __attribute__((ext_vector_type(8)))  float    v8f;

typedef int i32x4_ __attribute__((ext_vector_type(4)));
typedef __attribute__((address_space(1))) i32x4_ gvec128;   // global int4
typedef __attribute__((address_space(3))) i32x4_ lvec128;   // LDS int4

#define C_   256
#define CQ_  32
#define B_   8
#define N_   4096
#define INV_SQRT_CQ 0.17677669529663687f

// LDS chunk layout (per buffer): K = 64 rows x 40 halves (32 used + pad),
// V = 256 rows x 72 halves (64 used + pad). Double buffered.
#define KROW_ 40
#define VROW_ 72
#define KBUF_HALVES_ (64 * KROW_)     // 2560
#define VBUF_HALVES_ (256 * VROW_)    // 18432
#define VBASE_HALVES_ (2 * KBUF_HALVES_)
#define SMEM_HALVES_ (2 * KBUF_HALVES_ + 2 * VBUF_HALVES_)
#define SMEM_BYTES_  (SMEM_HALVES_ * 2)   // 83968 B

// ---------------------------------------------------------------------------
// CDNA5 async global->LDS copy (ASYNCcnt-tracked), with asm fallback.
// Probe-confirmed signature: (global int4*, local int4*, imm offset, imm cpol)
// ---------------------------------------------------------------------------
__device__ __forceinline__ void async_copy16(const _Float16* g, _Float16* l) {
#if __has_builtin(__builtin_amdgcn_global_load_async_to_lds_b128)
    __builtin_amdgcn_global_load_async_to_lds_b128(
        (gvec128*)g, (lvec128*)l, 0, 0);
#else
    asm volatile("global_load_async_to_lds_b128 %0, %1, off"
                 :: "v"((unsigned)(unsigned long long)l),
                    "v"((unsigned long long)g)
                 : "memory");
#endif
}

__device__ __forceinline__ void wait_async_le9() {
#if __has_builtin(__builtin_amdgcn_s_wait_asynccnt)
    __builtin_amdgcn_s_wait_asynccnt(9);
#else
    asm volatile("s_wait_asynccnt 9" ::: "memory");
#endif
}
__device__ __forceinline__ void wait_async_le0() {
#if __has_builtin(__builtin_amdgcn_s_wait_asynccnt)
    __builtin_amdgcn_s_wait_asynccnt(0);
#else
    asm volatile("s_wait_asynccnt 0" ::: "memory");
#endif
}

// Per-thread: 9 async b128 copies (1 K segment, 8 V segments = one V row).
__device__ __forceinline__ void load_kv_chunk(const _Float16* kb, const _Float16* vb,
                                              int nc, int tid,
                                              _Float16* Kc, _Float16* Vc) {
    const int row = tid >> 2, sg = tid & 3;          // K: 64 rows x 4 segs
    const _Float16* kg = kb + (size_t)nc * CQ_;
    async_copy16(kg + row * CQ_ + sg * 8, Kc + row * KROW_ + sg * 8);

    const _Float16* vg = vb + (size_t)tid * N_ + nc; // V row c = tid
    _Float16* vl = Vc + tid * VROW_;
    #pragma unroll
    for (int s = 0; s < 8; ++s) async_copy16(vg + s * 8, vl + s * 8);
}

// ---------------------------------------------------------------------------
// Kernel 1: Q/K/V projections via WMMA.
//   qh, kh : [B][N][32]  f16   (q pre-scaled by 1/sqrt(32))
//   vh     : [B][C][N]   f16
// ---------------------------------------------------------------------------
__global__ void __launch_bounds__(256) proj_kernel(
    const float* __restrict__ x,
    const float* __restrict__ wq, const float* __restrict__ wk,
    const float* __restrict__ wv,
    _Float16* __restrict__ qh, _Float16* __restrict__ kh,
    _Float16* __restrict__ vh)
{
    const int lane = threadIdx.x & 31;
    const int wave = threadIdx.x >> 5;
    const int b    = blockIdx.x >> 5;
    const int n0   = ((blockIdx.x & 31) * 8 + wave) * 16;
    const int g    = lane >> 4;
    const int lm   = lane & 15;

    const float* xb = x + (size_t)b * C_ * N_;

    // B operands (x -> f16) for all 8 K-steps; reused by all 20 o-tiles.
    v16h Bop[8];
    for (int k = 0; k < 8; ++k) {
        const int cbase = 32 * k + (g ? 16 : 0);
        #pragma unroll
        for (int i = 0; i < 16; ++i)
            Bop[k][i] = (_Float16)xb[(size_t)(cbase + i) * N_ + (n0 + lm)];
    }

    const int goff = g ? 8 : 0;
    for (int ot = 0; ot < 20; ++ot) {
        const float* wbase;
        int o0;
        if (ot < 2)      { wbase = wq; o0 = 16 * ot; }
        else if (ot < 4) { wbase = wk; o0 = 16 * (ot - 2); }
        else             { wbase = wv; o0 = 16 * (ot - 4); }

        v8f acc = {};
        for (int k = 0; k < 8; ++k) {
            const float* p = wbase + (size_t)(o0 + lm) * C_ + 32 * k;
            v16h A;
            #pragma unroll
            for (int i = 0; i < 8; ++i) {
                A[i]     = (_Float16)p[2 * goff + i];
                A[8 + i] = (_Float16)p[2 * goff + 8 + i];
            }
            acc = __builtin_amdgcn_wmma_f32_16x16x32_f16(
                      false, A, false, Bop[k], (short)0, acc, false, false);
        }

        if (ot < 4) {
            const bool  isq   = (ot < 2);
            const float scale = isq ? INV_SQRT_CQ : 1.0f;
            _Float16* dst = (isq ? qh : kh)
                          + ((size_t)b * N_ + (n0 + lm)) * CQ_ + o0 + goff;
            v8h h;
            #pragma unroll
            for (int r = 0; r < 8; ++r) h[r] = (_Float16)(acc[r] * scale);
            *(v8h*)dst = h;
        } else {
            _Float16* dst = vh + (size_t)b * C_ * N_;
            #pragma unroll
            for (int r = 0; r < 8; ++r) {
                const int cc = o0 + goff + r;
                dst[(size_t)cc * N_ + (n0 + lm)] = (_Float16)acc[r];
            }
        }
    }
}

// ---------------------------------------------------------------------------
// Kernel 2: flash-attention with block-shared, double-buffered, async-loaded
// K/V chunks in LDS. One wave owns 16 query rows + full C=256 output.
// ---------------------------------------------------------------------------
__global__ void __launch_bounds__(256) attn_kernel(
    const float* __restrict__ x, const float* __restrict__ gamma,
    const _Float16* __restrict__ qh, const _Float16* __restrict__ kh,
    const _Float16* __restrict__ vh, float* __restrict__ out)
{
    extern __shared__ _Float16 smem[];

    const int tid  = threadIdx.x;
    const int lane = tid & 31;
    const int wave = tid >> 5;
    const int b    = blockIdx.x >> 5;
    const int m0   = ((blockIdx.x & 31) * 8 + wave) * 16;
    const int g    = lane >> 4;
    const int lm   = lane & 15;
    const int goff = g ? 8 : 0;

    const _Float16* qb = qh + (size_t)b * N_ * CQ_;
    const _Float16* kb = kh + (size_t)b * N_ * CQ_;
    const _Float16* vb = vh + (size_t)b * C_ * N_;

    // Q as B-operand (fixed): lane -> m=m0+lm ; half i -> K(o)=(g?16:0)+i
    v16h Bq = *(const v16h*)(qb + (size_t)(m0 + lm) * CQ_ + (g ? 16 : 0));

    v8f acc[16];
    #pragma unroll
    for (int t = 0; t < 16; ++t) acc[t] = (v8f){};

    float run_max = -1e30f;
    float run_sum = 0.0f;

    // prologue: chunk 0 -> buffer 0
    load_kv_chunk(kb, vb, 0, tid, smem, smem + VBASE_HALVES_);

    const int NCHUNK = N_ / 64;
    for (int ic = 0; ic < NCHUNK; ++ic) {
        const int cur = ic & 1;
        const _Float16* Kc = smem + cur * KBUF_HALVES_;
        const _Float16* Vc = smem + VBASE_HALVES_ + cur * VBUF_HALVES_;

        if (ic + 1 < NCHUNK) {
            const int nxt = (ic + 1) & 1;
            load_kv_chunk(kb, vb, (ic + 1) * 64, tid,
                          smem + nxt * KBUF_HALVES_,
                          smem + VBASE_HALVES_ + nxt * VBUF_HALVES_);
            wait_async_le9();   // previous chunk's 9 copies done; next 9 in flight
        } else {
            wait_async_le0();
        }
        __syncthreads();        // all waves' LDS writes for this chunk visible

        // ---- scores^T: 4 tiles, D[n_local][m] = sum_o k[o,n]*q_scaled[o,m]
        v8f S[4];
        #pragma unroll
        for (int s = 0; s < 4; ++s) {
            const _Float16* p = Kc + (16 * s + lm) * KROW_;
            v8h lo = *(const v8h*)(p + 2 * goff);
            v8h hi = *(const v8h*)(p + 2 * goff + 8);
            v16h A;
            #pragma unroll
            for (int i = 0; i < 8; ++i) { A[i] = lo[i]; A[8 + i] = hi[i]; }
            v8f z = {};
            S[s] = __builtin_amdgcn_wmma_f32_16x16x32_f16(
                       false, A, false, Bq, (short)0, z, false, false);
        }

        // ---- online softmax for column m (this lane + partner lane l^16)
        float cmax = -1e30f;
        #pragma unroll
        for (int s = 0; s < 4; ++s)
            #pragma unroll
            for (int r = 0; r < 8; ++r) cmax = fmaxf(cmax, S[s][r]);
        cmax = fmaxf(cmax, __shfl_xor(cmax, 16, 32));
        const float nmax  = fmaxf(run_max, cmax);
        const float alpha = __expf(run_max - nmax);
        run_max = nmax;

        float P[4][8];
        float psum = 0.0f;
        #pragma unroll
        for (int s = 0; s < 4; ++s)
            #pragma unroll
            for (int r = 0; r < 8; ++r) {
                const float pv = __expf(S[s][r] - nmax);
                P[s][r] = pv;
                psum += pv;
            }
        run_sum = run_sum * alpha + psum;

        float Po[4][8];
        #pragma unroll
        for (int s = 0; s < 4; ++s)
            #pragma unroll
            for (int r = 0; r < 8; ++r) Po[s][r] = __shfl_xor(P[s][r], 16, 32);

        // ---- P^T as two B-operands (K = n_local)
        v16h B0, B1;
        #pragma unroll
        for (int i = 0; i < 8; ++i) {
            B0[i]     = (_Float16)(g ? Po[1][i] : P[0][i]);
            B0[8 + i] = (_Float16)(g ? P[1][i]  : Po[0][i]);
            B1[i]     = (_Float16)(g ? Po[3][i] : P[2][i]);
            B1[8 + i] = (_Float16)(g ? P[3][i]  : Po[2][i]);
        }

        // ---- Out^T[c,m] = alpha*Out^T + V(16c x 64n) * P^T(64n x 16m)
        for (int t = 0; t < 16; ++t) {
            v8f a = acc[t];
            #pragma unroll
            for (int r = 0; r < 8; ++r) a[r] *= alpha;

            const _Float16* vp = Vc + (16 * t + lm) * VROW_;
            v16h A0, A1;
            {
                v8h l0 = *(const v8h*)(vp + 2 * goff);
                v8h h0 = *(const v8h*)(vp + 2 * goff + 8);
                v8h l1 = *(const v8h*)(vp + 32 + 2 * goff);
                v8h h1 = *(const v8h*)(vp + 40 + 2 * goff);
                #pragma unroll
                for (int i = 0; i < 8; ++i) {
                    A0[i] = l0[i]; A0[8 + i] = h0[i];
                    A1[i] = l1[i]; A1[8 + i] = h1[i];
                }
            }
            a = __builtin_amdgcn_wmma_f32_16x16x32_f16(
                    false, A0, false, B0, (short)0, a, false, false);
            a = __builtin_amdgcn_wmma_f32_16x16x32_f16(
                    false, A1, false, B1, (short)0, a, false, false);
            acc[t] = a;
        }

        __syncthreads();   // all reads done before buffer is refilled (ic+2)
    }

    // ---- finalize: normalize, gamma * out + x
    const float tot = run_sum + __shfl_xor(run_sum, 16, 32);
    const float inv = 1.0f / tot;
    const float gam = gamma[0];
    const float* xb = x   + (size_t)b * C_ * N_;
    float*       ob = out + (size_t)b * C_ * N_;
    const int m = m0 + lm;
    for (int t = 0; t < 16; ++t) {
        #pragma unroll
        for (int r = 0; r < 8; ++r) {
            const int    cc  = 16 * t + goff + r;
            const size_t idx = (size_t)cc * N_ + m;
            ob[idx] = gam * (acc[t][r] * inv) + xb[idx];
        }
    }
}

// ---------------------------------------------------------------------------
extern "C" void kernel_launch(void* const* d_in, const int* in_sizes, int n_in,
                              void* d_out, int out_size, void* d_ws, size_t ws_size,
                              hipStream_t stream) {
    const float* x     = (const float*)d_in[0];
    const float* wq    = (const float*)d_in[1];
    const float* wk    = (const float*)d_in[2];
    const float* wv    = (const float*)d_in[3];
    const float* gamma = (const float*)d_in[4];
    float*       out   = (float*)d_out;

    _Float16* qh = (_Float16*)d_ws;                       // 2 MB
    _Float16* kh = qh + (size_t)B_ * N_ * CQ_;            // 2 MB
    _Float16* vh = kh + (size_t)B_ * N_ * CQ_;            // 16 MB

    proj_kernel<<<256, 256, 0, stream>>>(x, wq, wk, wv, qh, kh, vh);
    attn_kernel<<<256, 256, SMEM_BYTES_, stream>>>(x, gamma, qh, kh, vh, out);
}
